// MOELayer_69604239999273
// MI455X (gfx1250) — matrix operand order repaired and checked by reading, not compiled, not run on hardware
//
#include <hip/hip_runtime.h>
#include <math.h>

#define D_DIM 2048
#define I_DIM 8192
#define E_NUM 8
#define N_TOK 4096
#define GATE_H 64

// ---- device-only bf16 vector types -----------------------------------------
typedef __bf16 v16bf __attribute__((ext_vector_type(16)));
typedef __bf16 v8bf  __attribute__((ext_vector_type(8)));
typedef float  v8f   __attribute__((ext_vector_type(8)));
typedef int    v4i   __attribute__((vector_size(16)));       // matches builtin param

union FragBF { v16bf v; v8bf h[2]; };

static __device__ __forceinline__ __bf16 f2bf(float f) {
  union { float f; unsigned u; } in; in.f = f;
  unsigned u = (in.u + 0x7FFFu + ((in.u >> 16) & 1u)) >> 16; // RNE
  union { unsigned short s; __bf16 b; } out; out.s = (unsigned short)u;
  return out.b;
}

// ---- CDNA5 async global->LDS copy (guarded; falls back to plain copy) ------
#if __has_builtin(__builtin_amdgcn_global_load_async_to_lds_b128)
#define HAVE_ASYNC_LDS 1
typedef __attribute__((address_space(1))) v4i* gas_v4i_ptr;  // global (device) AS
typedef __attribute__((address_space(3))) v4i* las_v4i_ptr;  // LDS (shared) AS
#else
#define HAVE_ASYNC_LDS 0
#endif

static __device__ __forceinline__ void async_wait0() {
#if __has_builtin(__builtin_amdgcn_s_wait_asynccnt)
  __builtin_amdgcn_s_wait_asynccnt(0);
#elif HAVE_ASYNC_LDS
  asm volatile("s_wait_asynccnt 0x0" ::: "memory");
#endif
}

// ---------------------------------------------------------------------------
// Kernel 1: router. One 256-thread block per token.
//   outputs: logits[N,8] into d_out tail, rs[N,8] = {w_copy, w_ce0, w_ce1,
//            w_swiglu, cw00, cw01, cw10, cw11}
// ---------------------------------------------------------------------------
__launch_bounds__(256)
__global__ void router_kernel(const float* __restrict__ x,
                              const float* __restrict__ gw1,   // [D,64]
                              const float* __restrict__ gw2,   // [64,8]
                              const float* __restrict__ cewg,  // [2,D,2]
                              float* __restrict__ logits_out,  // [N,8]
                              float* __restrict__ rs)          // [N,8]
{
  const int tok = blockIdx.x;
  const int t = threadIdx.x;
  const float* xr = x + (size_t)tok * D_DIM;

  __shared__ float xs[D_DIM];
  __shared__ float red[256];
  __shared__ float t1[GATE_H];
  __shared__ float cred[4][256];
  __shared__ float sc[8];

  for (int i = t; i < D_DIM; i += 256) xs[i] = xr[i];
  __syncthreads();

  // const-expert gate dot products (4 x len-2048)
  float c0a = 0.f, c0b = 0.f, c1a = 0.f, c1b = 0.f;
  for (int k = t; k < D_DIM; k += 256) {
    float xv = xs[k];
    c0a += xv * cewg[(0 * D_DIM + k) * 2 + 0];
    c0b += xv * cewg[(0 * D_DIM + k) * 2 + 1];
    c1a += xv * cewg[(1 * D_DIM + k) * 2 + 0];
    c1b += xv * cewg[(1 * D_DIM + k) * 2 + 1];
  }
  cred[0][t] = c0a; cred[1][t] = c0b; cred[2][t] = c1a; cred[3][t] = c1b;

  // gate_w1: column j = t&63, k-slice g = t>>6
  const int j = t & 63, g = t >> 6;
  float s = 0.f;
  const int k0 = g * (D_DIM / 4);
  for (int k = k0; k < k0 + D_DIM / 4; ++k) s += xs[k] * gw1[k * GATE_H + j];
  red[t] = s;
  __syncthreads();

  if (t < GATE_H) {
    float v = red[j] + red[64 + j] + red[128 + j] + red[192 + j];
    t1[j] = tanhf(v);
  }
  // reduce the 4 const-expert partials
  for (int off = 128; off >= 1; off >>= 1) {
    __syncthreads();
    if (t < off) {
      cred[0][t] += cred[0][t + off]; cred[1][t] += cred[1][t + off];
      cred[2][t] += cred[2][t + off]; cred[3][t] += cred[3][t + off];
    }
  }
  __syncthreads();

  if (t < E_NUM) {
    float l = 0.f;
    for (int jj = 0; jj < GATE_H; ++jj) l += t1[jj] * gw2[jj * E_NUM + t];
    sc[t] = l;
    logits_out[tok * E_NUM + t] = l;
  }
  __syncthreads();

  if (t == 0) {
    float p[E_NUM];
    float mx = sc[0];
    for (int e = 1; e < E_NUM; ++e) mx = fmaxf(mx, sc[e]);
    float se = 0.f;
    for (int e = 0; e < E_NUM; ++e) { p[e] = __expf(sc[e] - mx); se += p[e]; }
    for (int e = 0; e < E_NUM; ++e) p[e] /= se;
    // top-2 (earliest index wins ties, like lax.top_k)
    int i0 = 0;
    for (int e = 1; e < E_NUM; ++e) if (p[e] > p[i0]) i0 = e;
    int i1 = (i0 == 0) ? 1 : 0;
    for (int e = 0; e < E_NUM; ++e) {
      if (e == i0) continue;
      if (p[e] > p[i1]) i1 = e;
    }
    float w0 = (i0 == E_NUM - 1) ? 0.f : p[i0];
    float w1 = (i1 == E_NUM - 1) ? 0.f : p[i1];
    float wsum = w0 + w1;
    w0 /= wsum; w1 /= wsum;
    float pe[E_NUM];
    for (int e = 0; e < E_NUM; ++e) pe[e] = 0.f;
    pe[i0] += w0; pe[i1] += w1;
    float wsw = pe[4] + pe[5] + pe[6] + pe[7];

    float a0 = cred[0][0], b0 = cred[1][0], a1 = cred[2][0], b1 = cred[3][0];
    float m0 = fmaxf(a0, b0), m1 = fmaxf(a1, b1);
    float e0a = __expf(a0 - m0), e0b = __expf(b0 - m0), s0 = e0a + e0b;
    float e1a = __expf(a1 - m1), e1b = __expf(b1 - m1), s1 = e1a + e1b;

    float* r = rs + tok * 8;
    r[0] = pe[0]; r[1] = pe[2]; r[2] = pe[3]; r[3] = wsw;
    r[4] = e0a / s0; r[5] = e0b / s0; r[6] = e1a / s1; r[7] = e1b / s1;
  }
}

// ---------------------------------------------------------------------------
// Kernel 2: base = w_copy*x + w_ce0*(cw00*x + cw01*c0) + w_ce1*(cw10*x + cw11*c1)
// ---------------------------------------------------------------------------
__launch_bounds__(256)
__global__ void base_kernel(const float* __restrict__ x,
                            const float* __restrict__ cec,   // [2,D]
                            const float* __restrict__ rs,
                            float* __restrict__ out)
{
  int idx = blockIdx.x * 256 + threadIdx.x;
  size_t base = (size_t)idx * 4;
  int tok = (int)(base / D_DIM);
  int d0 = (int)(base % D_DIM);
  const float* r = rs + tok * 8;
  float wce0 = r[1], wce1 = r[2];
  float xf = r[0] + wce0 * r[4] + wce1 * r[6];
  float k0 = wce0 * r[5], k1 = wce1 * r[7];
  float4 xv = *(const float4*)(x + base);
  float4 c0 = *(const float4*)(cec + d0);
  float4 c1 = *(const float4*)(cec + D_DIM + d0);
  float4 o;
  o.x = xf * xv.x + k0 * c0.x + k1 * c1.x;
  o.y = xf * xv.y + k0 * c0.y + k1 * c1.y;
  o.z = xf * xv.z + k0 * c0.z + k1 * c1.z;
  o.w = xf * xv.w + k0 * c0.w + k1 * c1.w;
  *(float4*)(out + base) = o;
}

// ---------------------------------------------------------------------------
// Kernel 3: act[N,I] = bf16( silu(x@w_gate) * (x@w_up) )
// Block tile 128(M) x 64(N), BK=32, 256 threads = 8 waves in 4x2 layout.
// ---------------------------------------------------------------------------
__launch_bounds__(256)
__global__ void gemm1_swiglu(const float* __restrict__ x,    // [N,D]
                             const float* __restrict__ wg,   // [D,I]
                             const float* __restrict__ wu,   // [D,I]
                             unsigned short* __restrict__ act_raw)
{
  __bf16* __restrict__ act = reinterpret_cast<__bf16*>(act_raw);
  const int nb = blockIdx.x * 64;
  const int mb = blockIdx.y * 128;
  const int t = threadIdx.x;
  const int lane = t & 31, wid = t >> 5;
  const int waveM = wid >> 1, waveN = wid & 1;

  __shared__ __align__(16) __bf16 As[128 * 40];
  __shared__ __align__(16) __bf16 Bg[64 * 40];
  __shared__ __align__(16) __bf16 Bu[64 * 40];

  const v8f vzero = {0.f, 0.f, 0.f, 0.f, 0.f, 0.f, 0.f, 0.f};
  v8f accg[2][2], accu[2][2];
  for (int mi = 0; mi < 2; ++mi)
    for (int ni = 0; ni < 2; ++ni) { accg[mi][ni] = vzero; accu[mi][ni] = vzero; }

  const int arow = (waveM * 32 + (lane & 15)) * 40;
  const int khalf = (lane >> 4) * 16;    // A: K run kh..kh+15
  const int kbase = (lane >> 4) * 16;    // B: K run kb..kb+15

  for (int kb = 0; kb < D_DIM; kb += 32) {
    // stage A (128x32 fp32 -> bf16), float4 global loads
    #pragma unroll
    for (int i = 0; i < 4; ++i) {
      int c = t + i * 256;
      int row = c >> 3, kc = (c & 7) << 2;
      const float* src = x + (size_t)(mb + row) * D_DIM + kb + kc;
      float4 v = *(const float4*)src;
      if (kb + 32 < D_DIM) __builtin_prefetch(src + 32, 0, 1);
      __bf16* dst = As + row * 40 + kc;
      dst[0] = f2bf(v.x); dst[1] = f2bf(v.y); dst[2] = f2bf(v.z); dst[3] = f2bf(v.w);
    }
    // stage B tiles (gate & up), transposed to [n][k] in LDS
    #pragma unroll
    for (int i = 0; i < 2; ++i) {
      int c = t + i * 256;
      int k = c >> 4, nc = (c & 15) << 2;
      const float* sg = wg + (size_t)(kb + k) * I_DIM + nb + nc;
      const float* su = wu + (size_t)(kb + k) * I_DIM + nb + nc;
      float4 vg = *(const float4*)sg;
      float4 vu = *(const float4*)su;
      if (kb + 32 < D_DIM) {
        __builtin_prefetch(sg + (size_t)32 * I_DIM, 0, 1);
        __builtin_prefetch(su + (size_t)32 * I_DIM, 0, 1);
      }
      Bg[(nc + 0) * 40 + k] = f2bf(vg.x); Bg[(nc + 1) * 40 + k] = f2bf(vg.y);
      Bg[(nc + 2) * 40 + k] = f2bf(vg.z); Bg[(nc + 3) * 40 + k] = f2bf(vg.w);
      Bu[(nc + 0) * 40 + k] = f2bf(vu.x); Bu[(nc + 1) * 40 + k] = f2bf(vu.y);
      Bu[(nc + 2) * 40 + k] = f2bf(vu.z); Bu[(nc + 3) * 40 + k] = f2bf(vu.w);
    }
    __syncthreads();

    FragBF a[2], bg[2], bu[2];
    #pragma unroll
    for (int mi = 0; mi < 2; ++mi) {
      const __bf16* p = As + arow + mi * 16 * 40;
      a[mi].h[0] = *(const v8bf*)(p + khalf);
      a[mi].h[1] = *(const v8bf*)(p + 8 + khalf);
    }
    #pragma unroll
    for (int ni = 0; ni < 2; ++ni) {
      int col = waveN * 32 + ni * 16 + (lane & 15);
      const __bf16* pg = Bg + col * 40 + kbase;
      const __bf16* pu = Bu + col * 40 + kbase;
      bg[ni].h[0] = *(const v8bf*)pg;        bg[ni].h[1] = *(const v8bf*)(pg + 8);
      bu[ni].h[0] = *(const v8bf*)pu;        bu[ni].h[1] = *(const v8bf*)(pu + 8);
    }
    #pragma unroll
    for (int mi = 0; mi < 2; ++mi)
      #pragma unroll
      for (int ni = 0; ni < 2; ++ni) {
        accg[mi][ni] = __builtin_amdgcn_wmma_f32_16x16x32_bf16(
            false, a[mi].v, false, bg[ni].v, (short)0, accg[mi][ni], false, false);
        accu[mi][ni] = __builtin_amdgcn_wmma_f32_16x16x32_bf16(
            false, a[mi].v, false, bu[ni].v, (short)0, accu[mi][ni], false, false);
      }
    __syncthreads();
  }

  // epilogue: silu(g)*u -> bf16 act  (fast rcp instead of IEEE divide)
  const int lm = ((lane >> 4) & 1) * 8;
  const int ln = lane & 15;
  #pragma unroll
  for (int mi = 0; mi < 2; ++mi)
    #pragma unroll
    for (int ni = 0; ni < 2; ++ni)
      #pragma unroll
      for (int r = 0; r < 8; ++r) {
        float gv = accg[mi][ni][r];
        float uv = accu[mi][ni][r];
        float sig = __builtin_amdgcn_rcpf(1.f + __expf(-gv));
        float hv = gv * sig * uv;
        int row = mb + waveM * 32 + mi * 16 + lm + r;
        int col = nb + waveN * 32 + ni * 16 + ln;
        act[(size_t)row * I_DIM + col] = f2bf(hv);
      }
}

// ---------------------------------------------------------------------------
// Kernel 4: out[t,d] += w_swiglu[t] * (act @ w_down)[t,d]
// A-tile staging uses CDNA5 async global->LDS copies when available.
// ---------------------------------------------------------------------------
__launch_bounds__(256)
__global__ void gemm2_down(const unsigned short* __restrict__ act_raw, // [N,I] bf16
                           const float* __restrict__ wd,               // [I,D]
                           const float* __restrict__ rs,
                           float* __restrict__ out)                    // [N,D]
{
  const __bf16* __restrict__ act = reinterpret_cast<const __bf16*>(act_raw);
  const int nb = blockIdx.x * 64;   // over D
  const int mb = blockIdx.y * 128;  // over tokens
  const int t = threadIdx.x;
  const int lane = t & 31, wid = t >> 5;
  const int waveM = wid >> 1, waveN = wid & 1;

  __shared__ __align__(16) __bf16 As[128 * 40];
  __shared__ __align__(16) __bf16 Bs[64 * 40];

  const v8f vzero = {0.f, 0.f, 0.f, 0.f, 0.f, 0.f, 0.f, 0.f};
  v8f acc[2][2];
  for (int mi = 0; mi < 2; ++mi)
    for (int ni = 0; ni < 2; ++ni) acc[mi][ni] = vzero;

  const int arow = (waveM * 32 + (lane & 15)) * 40;
  const int khalf = (lane >> 4) * 16;
  const int kbase = (lane >> 4) * 16;

  for (int kb = 0; kb < I_DIM; kb += 32) {
    // stage A: bf16 activations, 16B chunks (async DMA to LDS when available)
    #pragma unroll
    for (int i = 0; i < 2; ++i) {
      int c = t + i * 256;
      int row = c >> 2, kc = (c & 3) << 3;
      const __bf16* gsrc = act + (size_t)(mb + row) * I_DIM + kb + kc;
      __bf16* ldst = As + row * 40 + kc;
#if HAVE_ASYNC_LDS
      __builtin_amdgcn_global_load_async_to_lds_b128(
          (gas_v4i_ptr)gsrc, (las_v4i_ptr)ldst, 0, 0);
#else
      *(v8bf*)ldst = *(const v8bf*)gsrc;
#endif
      if (kb + 32 < I_DIM) __builtin_prefetch(gsrc + 32, 0, 1);
    }
    // stage B: w_down fp32 -> bf16, transposed to [n][k]
    #pragma unroll
    for (int i = 0; i < 2; ++i) {
      int c = t + i * 256;
      int k = c >> 4, nc = (c & 15) << 2;
      const float* src = wd + (size_t)(kb + k) * D_DIM + nb + nc;
      float4 v = *(const float4*)src;
      if (kb + 32 < I_DIM) __builtin_prefetch(src + (size_t)32 * D_DIM, 0, 1);
      Bs[(nc + 0) * 40 + k] = f2bf(v.x); Bs[(nc + 1) * 40 + k] = f2bf(v.y);
      Bs[(nc + 2) * 40 + k] = f2bf(v.z); Bs[(nc + 3) * 40 + k] = f2bf(v.w);
    }
#if HAVE_ASYNC_LDS
    async_wait0();
#endif
    __syncthreads();

    FragBF a[2], b[2];
    #pragma unroll
    for (int mi = 0; mi < 2; ++mi) {
      const __bf16* p = As + arow + mi * 16 * 40;
      a[mi].h[0] = *(const v8bf*)(p + khalf);
      a[mi].h[1] = *(const v8bf*)(p + 8 + khalf);
    }
    #pragma unroll
    for (int ni = 0; ni < 2; ++ni) {
      int col = waveN * 32 + ni * 16 + (lane & 15);
      const __bf16* pb = Bs + col * 40 + kbase;
      b[ni].h[0] = *(const v8bf*)pb;
      b[ni].h[1] = *(const v8bf*)(pb + 8);
    }
    #pragma unroll
    for (int mi = 0; mi < 2; ++mi)
      #pragma unroll
      for (int ni = 0; ni < 2; ++ni)
        acc[mi][ni] = __builtin_amdgcn_wmma_f32_16x16x32_bf16(
            false, a[mi].v, false, b[ni].v, (short)0, acc[mi][ni], false, false);
    __syncthreads();
  }

  const int lm = ((lane >> 4) & 1) * 8;
  const int ln = lane & 15;
  #pragma unroll
  for (int mi = 0; mi < 2; ++mi)
    #pragma unroll
    for (int ni = 0; ni < 2; ++ni)
      #pragma unroll
      for (int r = 0; r < 8; ++r) {
        int row = mb + waveM * 32 + mi * 16 + lm + r;
        int col = nb + waveN * 32 + ni * 16 + ln;
        float wsw = rs[row * 8 + 3];
        out[(size_t)row * D_DIM + col] += wsw * acc[mi][ni][r];
      }
}

// ---------------------------------------------------------------------------
extern "C" void kernel_launch(void* const* d_in, const int* in_sizes, int n_in,
                              void* d_out, int out_size, void* d_ws, size_t ws_size,
                              hipStream_t stream) {
  const float* x     = (const float*)d_in[0];  // [2,2048,2048]
  const float* gw1   = (const float*)d_in[1];  // [2048,64]
  const float* gw2   = (const float*)d_in[2];  // [64,8]
  const float* cewg  = (const float*)d_in[3];  // [2,2048,2]
  const float* cec   = (const float*)d_in[4];  // [2,2048]
  const float* wgate = (const float*)d_in[5];  // [2048,8192]
  const float* wup   = (const float*)d_in[6];  // [2048,8192]
  const float* wdown = (const float*)d_in[7];  // [8192,2048]

  float* out    = (float*)d_out;                       // [N,D]
  float* logits = out + (size_t)N_TOK * D_DIM;         // [N,8]

  float* rs = (float*)d_ws;                            // [N,8] routing scalars
  unsigned short* act =
      (unsigned short*)((char*)d_ws + 256 * 1024);     // [N,I] bf16 activations

  router_kernel<<<N_TOK, 256, 0, stream>>>(x, gw1, gw2, cewg, logits, rs);
  base_kernel<<<(N_TOK * (D_DIM / 4)) / 256, 256, 0, stream>>>(x, cec, rs, out);
  gemm1_swiglu<<<dim3(I_DIM / 64, N_TOK / 128), 256, 0, stream>>>(x, wgate, wup, act);
  gemm2_down<<<dim3(D_DIM / 64, N_TOK / 128), 256, 0, stream>>>(act, wdown, rs, out);
}
